// GRUCell_20203526160650
// MI455X (gfx1250) — matrix operand order repaired and checked by reading, not compiled, not run on hardware
//
#include <hip/hip_runtime.h>
#include <hip/hip_bf16.h>

typedef _Float16 v16h __attribute__((ext_vector_type(16)));
typedef _Float16 v8h  __attribute__((ext_vector_type(8)));
typedef _Float16 v4h  __attribute__((ext_vector_type(4)));
typedef float    v8f  __attribute__((ext_vector_type(8)));
typedef float    v4f  __attribute__((ext_vector_type(4)));

#define TM       128    // rows per block
#define KTOT     256    // D + U
#define NU       128    // output units
#define ASTRIDE  264    // f16 row stride for combined [s_prev|x] (256 + 8 pad)
#define WSTRIDE  264    // f16 row stride for transposed weights (256 + 8 pad)
#define RSTRIDE  136    // f16 row stride for r*s_prev buffer (128 + 8 pad)

// LDS carve (units: _Float16)
#define OFF_COMB 0
#define OFF_WZ   (TM * ASTRIDE)                 // 33792
#define OFF_WR   (OFF_WZ + NU * WSTRIDE)       // 67584
#define OFF_WS   (OFF_WR + NU * WSTRIDE)       // 101376
#define OFF_RS   (OFF_WS + NU * WSTRIDE)       // 135168
#define LDS_F16  (OFF_RS + TM * RSTRIDE)       // 152576 f16
#define LDS_BYTES (LDS_F16 * 2)                // 305152 B < 320KB

__device__ __forceinline__ v16h cat8(v8h lo, v8h hi) {
    return __builtin_shufflevector(lo, hi, 0,1,2,3,4,5,6,7,8,9,10,11,12,13,14,15);
}

// Load a 16x32 A fragment (f16) from an LDS row-major buffer.
// Layout: lane m = lane%16, half = lane/16; lane covers k = 16*half + 0..15
// contiguously (matches B fragment). Two ds_load_b128 per lane.
__device__ __forceinline__ v16h load_afrag(const _Float16* rowp, int half) {
    const _Float16* p = rowp + 16 * half;
    v8h lo = *(const v8h*)(p);
    v8h hi = *(const v8h*)(p + 8);
    return cat8(lo, hi);
}

// Load a 32x16 B fragment from transposed-weight LDS (Wt[n][k]).
// Layout: lane n = lane%16, half = lane/16; lane covers k = 16*half + 0..15
// contiguously -> two adjacent ds_load_b128.
__device__ __forceinline__ v16h load_bfrag(const _Float16* rowp) {
    v8h lo = *(const v8h*)(rowp);
    v8h hi = *(const v8h*)(rowp + 8);
    return cat8(lo, hi);
}

__device__ __forceinline__ void stage_weight(const float* __restrict__ W,
                                             _Float16* __restrict__ T, int tid) {
    // W is [KTOT][NU] row-major f32 -> T[n][k] f16 (transposed), coalesced b128 reads
    for (int i = tid; i < (KTOT * NU) / 4; i += 256) {
        int flat = i * 4;
        int k = flat >> 7;        // / NU
        int n = flat & (NU - 1);
        v4f wv = *(const v4f*)(W + flat);
        T[(n + 0) * WSTRIDE + k] = (_Float16)wv[0];
        T[(n + 1) * WSTRIDE + k] = (_Float16)wv[1];
        T[(n + 2) * WSTRIDE + k] = (_Float16)wv[2];
        T[(n + 3) * WSTRIDE + k] = (_Float16)wv[3];
    }
}

__global__ void __launch_bounds__(256)
gru_wmma_kernel(const float* __restrict__ x, const float* __restrict__ s_prev,
                const float* __restrict__ Wz, const float* __restrict__ bz,
                const float* __restrict__ Wr, const float* __restrict__ br,
                const float* __restrict__ Ws, const float* __restrict__ bs,
                float* __restrict__ out) {
    extern __shared__ _Float16 lds[];
    _Float16* comb = lds + OFF_COMB;   // [TM][ASTRIDE]  cols 0..127 = s_prev, 128..255 = x
    _Float16* wtZ  = lds + OFF_WZ;     // [NU][WSTRIDE]
    _Float16* wtR  = lds + OFF_WR;
    _Float16* wtS  = lds + OFF_WS;
    _Float16* rsb  = lds + OFF_RS;     // [TM][RSTRIDE]  r * s_prev (f16)

    const int tid = threadIdx.x;
    const long rowBase = (long)blockIdx.x * TM;

    // ---- stage weights (transposed f32->f16) ----
    stage_weight(Wz, wtZ, tid);
    stage_weight(Wr, wtR, tid);
    stage_weight(Ws, wtS, tid);

    // ---- stage combined = [s_prev | x] as f16, coalesced b128 reads ----
    for (int i = tid; i < (TM * 128) / 4; i += 256) {
        int flat = i * 4;
        int r = flat >> 7;
        int c = flat & 127;
        v4f sv = *(const v4f*)(s_prev + (rowBase + r) * 128 + c);
        v4f xv = *(const v4f*)(x      + (rowBase + r) * 128 + c);
        *(v4h*)(comb + r * ASTRIDE + c)       = __builtin_convertvector(sv, v4h);
        *(v4h*)(comb + r * ASTRIDE + 128 + c) = __builtin_convertvector(xv, v4h);
    }
    __syncthreads();

    const int w    = tid >> 5;     // wave id: row tile
    const int lane = tid & 31;
    const int l16  = lane & 15;
    const int half = lane >> 4;

    // ---- A fragments for the z/r GEMMs: full K=256 strip in registers ----
    v16h afrag[8];
    {
        const _Float16* arow = comb + (w * 16 + l16) * ASTRIDE;
#pragma unroll
        for (int t = 0; t < 8; ++t)
            afrag[t] = load_afrag(arow + 32 * t, half);
    }

    // ---- Phase A: z and r gates (shared A, two B streams) ----
    v8f zreg[8];
#pragma unroll
    for (int u = 0; u < 8; ++u) {
        v8f accZ = {};
        v8f accR = {};
        const _Float16* bzrow = wtZ + (u * 16 + l16) * WSTRIDE + 16 * half;
        const _Float16* brrow = wtR + (u * 16 + l16) * WSTRIDE + 16 * half;
#pragma unroll
        for (int t = 0; t < 8; ++t) {
            v16h bf = load_bfrag(bzrow + 32 * t);
            accZ = __builtin_amdgcn_wmma_f32_16x16x32_f16(
                false, afrag[t], false, bf, (short)0, accZ, false, false);
            v16h bf2 = load_bfrag(brrow + 32 * t);
            accR = __builtin_amdgcn_wmma_f32_16x16x32_f16(
                false, afrag[t], false, bf2, (short)0, accR, false, false);
        }
        const float bzv = bz[u * 16 + l16];
        const float brv = br[u * 16 + l16];
#pragma unroll
        for (int i = 0; i < 8; ++i) {
            // C/D layout: element (m = i + 8*half, n = l16) in VGPR i
            zreg[u][i] = 1.0f / (1.0f + __expf(-(accZ[i] + bzv)));
            float rv   = 1.0f / (1.0f + __expf(-(accR[i] + brv)));
            long  gr   = rowBase + w * 16 + i + 8 * half;
            float spv  = s_prev[gr * 128 + u * 16 + l16];    // f32 precision, L2-hot
            rsb[(w * 16 + i + 8 * half) * RSTRIDE + u * 16 + l16] =
                (_Float16)(rv * spv);
        }
    }
    __syncthreads();

    // ---- A fragments for candidate GEMM: [r*s_prev | x] ----
#pragma unroll
    for (int t = 0; t < 4; ++t)
        afrag[t] = load_afrag(rsb + (w * 16 + l16) * RSTRIDE + 32 * t, half);
#pragma unroll
    for (int t = 0; t < 4; ++t)
        afrag[4 + t] = load_afrag(comb + (w * 16 + l16) * ASTRIDE + 128 + 32 * t, half);

    // ---- Phase B: candidate state + blend + store ----
#pragma unroll
    for (int u = 0; u < 8; ++u) {
        v8f accS = {};
        const _Float16* bsrow = wtS + (u * 16 + l16) * WSTRIDE + 16 * half;
#pragma unroll
        for (int t = 0; t < 8; ++t) {
            v16h bf = load_bfrag(bsrow + 32 * t);
            accS = __builtin_amdgcn_wmma_f32_16x16x32_f16(
                false, afrag[t], false, bf, (short)0, accS, false, false);
        }
        const float bsv = bs[u * 16 + l16];
#pragma unroll
        for (int i = 0; i < 8; ++i) {
            float pre = accS[i] + bsv;
            float e   = __expf(-2.0f * pre);
            float sh  = (1.0f - e) / (1.0f + e);             // tanh
            long  gr  = rowBase + w * 16 + i + 8 * half;
            float spv = s_prev[gr * 128 + u * 16 + l16];
            float z   = zreg[u][i];
            out[gr * 128 + u * 16 + l16] = (1.0f - z) * spv + z * sh;
        }
    }
}

extern "C" void kernel_launch(void* const* d_in, const int* in_sizes, int n_in,
                              void* d_out, int out_size, void* d_ws, size_t ws_size,
                              hipStream_t stream) {
    const float* x      = (const float*)d_in[0];
    const float* s_prev = (const float*)d_in[1];
    const float* Wz     = (const float*)d_in[2];
    const float* bz     = (const float*)d_in[3];
    const float* Wr     = (const float*)d_in[4];
    const float* br     = (const float*)d_in[5];
    const float* Ws     = (const float*)d_in[6];
    const float* bs     = (const float*)d_in[7];
    float* out = (float*)d_out;

    const int B = in_sizes[0] / 128;        // 131072 rows
    const int grid = B / TM;                // 1024 blocks

    static_assert(LDS_BYTES <= 320 * 1024, "exceeds WGP LDS");
    (void)hipFuncSetAttribute((const void*)gru_wmma_kernel,
                              hipFuncAttributeMaxDynamicSharedMemorySize,
                              LDS_BYTES);

    gru_wmma_kernel<<<grid, 256, LDS_BYTES, stream>>>(
        x, s_prev, Wz, bz, Wr, br, Ws, bs, out);
}
